// MambaBlock_87866440941885
// MI455X (gfx1250) — matrix operand order repaired and checked by reading, not compiled, not run on hardware
//
#include <hip/hip_runtime.h>

// ---------------------------------------------------------------------------
// Mamba-style block for MI455X (gfx1250, wave32, WMMA).
//   B=4, L=4096, D=1024.
// Pipeline:
//   1) layernorm  : X -> Zbf16                          (bf16 for WMMA)
//   2) wconv      : Wa|Wb|Wc f32 -> Wbf16 [3*D, D]
//   3) gemm3      : G[b*l, 0:1024)=sigmoid(Z Wa^T+ba), [1024:2048)=Z Wb^T+bb,
//                   [2048:3072)=Z Wc^T+bc   via v_wmma_f32_16x16x32_bf16,
//                   2x2 register-blocked (32x32 per wave)
//   4) chunked scan (2-level, 32 chunks x 128 steps) replicating the reference
//      P=cumprod(A), S=cumsum(Bx/(P+eps)), h=P*(h0+S), out=C*h+X
// ---------------------------------------------------------------------------

typedef __bf16 v16bf __attribute__((ext_vector_type(16)));
typedef __bf16 v8bf  __attribute__((ext_vector_type(8)));
typedef __bf16 v4bf  __attribute__((ext_vector_type(4)));
typedef float  v8f   __attribute__((ext_vector_type(8)));

#define BB   4
#define LL   4096
#define DD   1024
#define D3   3072
#define NCH  32            // chunks along L
#define LCH  128           // steps per chunk
#define LN_EPS   1e-5f
#define SCAN_EPS 1e-8f

// ---------------- 1) LayerNorm + bf16 convert -------------------------------
__global__ __launch_bounds__(256)
void mamba_ln_kernel(const float* __restrict__ X,
                     const float* __restrict__ gamma,
                     const float* __restrict__ beta,
                     __bf16* __restrict__ Zbf)
{
    const int row = blockIdx.x;               // 0 .. B*L-1
    const int tid = threadIdx.x;              // 256 threads, 4 elems each
    const float4 x = *(const float4*)&X[(size_t)row * DD + tid * 4];

    __shared__ float rs[256];
    __shared__ float rq[256];
    rs[tid] = x.x + x.y + x.z + x.w;
    rq[tid] = x.x * x.x + x.y * x.y + x.z * x.z + x.w * x.w;
    __syncthreads();
    for (int off = 128; off > 0; off >>= 1) {
        if (tid < off) { rs[tid] += rs[tid + off]; rq[tid] += rq[tid + off]; }
        __syncthreads();
    }
    const float mu  = rs[0] * (1.0f / DD);
    const float var = rq[0] * (1.0f / DD) - mu * mu;
    const float inv = rsqrtf(var + LN_EPS);

    const float4 g = *(const float4*)&gamma[tid * 4];
    const float4 b = *(const float4*)&beta[tid * 4];
    v4bf z;
    z[0] = (__bf16)((x.x - mu) * inv * g.x + b.x);
    z[1] = (__bf16)((x.y - mu) * inv * g.y + b.y);
    z[2] = (__bf16)((x.z - mu) * inv * g.z + b.z);
    z[3] = (__bf16)((x.w - mu) * inv * g.w + b.w);
    *(v4bf*)&Zbf[(size_t)row * DD + tid * 4] = z;
}

// ---------------- 2) weight f32 -> bf16 ------------------------------------
__global__ __launch_bounds__(256)
void mamba_wconv_kernel(const float* __restrict__ Wa,
                        const float* __restrict__ Wb,
                        const float* __restrict__ Wc,
                        __bf16* __restrict__ Wbf)
{
    const int i4 = blockIdx.x * blockDim.x + threadIdx.x;   // 4 floats each
    const size_t base = (size_t)i4 * 4;
    const int sec = (int)(base / (DD * DD));                // 0,1,2
    const size_t off = base - (size_t)sec * DD * DD;
    const float* src = (sec == 0) ? Wa : (sec == 1) ? Wb : Wc;
    const float4 w = *(const float4*)&src[off];
    v4bf o;
    o[0] = (__bf16)w.x; o[1] = (__bf16)w.y; o[2] = (__bf16)w.z; o[3] = (__bf16)w.w;
    *(v4bf*)&Wbf[base] = o;
}

// ---------------- 3) fused triple GEMM via WMMA -----------------------------
// grid: (3072/256, (B*L)/32). block: 256 = 8 waves; each wave a 32x32 patch
// (2x2 tiles of 16x16), K swept in steps of 32 with v_wmma_f32_16x16x32_bf16.
#define ZPITCH 1032      // 1024 + 8 bf16 pad -> 16B row skew, bank-conflict free
#define GEMM_LDS_BYTES (32 * ZPITCH * 2)

__global__ __launch_bounds__(256)
void mamba_gemm3_kernel(const __bf16* __restrict__ Z,
                        const __bf16* __restrict__ W,
                        const float* __restrict__ ba,
                        const float* __restrict__ bb,
                        const float* __restrict__ bc,
                        float* __restrict__ G)
{
    extern __shared__ __align__(16) __bf16 zs[];          // 32 * ZPITCH bf16

    const int rowBase = blockIdx.y * 32;
    const int wave    = threadIdx.x >> 5;
    const int colBase = blockIdx.x * 256 + wave * 32;     // 32 cols per wave
    const int t       = threadIdx.x;

    // stage 32x1024 bf16 Z strip: 4096 16-byte chunks, 16 per thread
    for (int ch = t; ch < 32 * 128; ch += 256) {
        const int r  = ch >> 7;
        const int c8 = (ch & 127) * 8;
        *(v8bf*)&zs[r * ZPITCH + c8] =
            *(const v8bf*)&Z[(size_t)(rowBase + r) * DD + c8];
    }
    __syncthreads();

    const int lane = t & 31;
    const int half = lane >> 4;       // 0 | 1
    const int mn   = lane & 15;       // row (A) / col (B) within tile

    union Frag { v16bf v; struct { v8bf lo; v8bf hi; } s; };

    v8f acc00 = {}, acc01 = {}, acc10 = {}, acc11 = {};
    const __bf16* wrow0 = W + (size_t)(colBase + mn) * DD;        // N tile 0
    const __bf16* wrow1 = W + (size_t)(colBase + 16 + mn) * DD;   // N tile 1

    for (int k0 = 0; k0 < DD; k0 += 32) {
        Frag a0, a1, b0, b1;
        // A 16x32 bf16 ISA layout: elems 0..7 -> K=h*8.., elems 8..15 -> K=16+h*8..
        a0.s.lo = *(const v8bf*)&zs[mn * ZPITCH + k0 + half * 8];
        a0.s.hi = *(const v8bf*)&zs[mn * ZPITCH + k0 + 16 + half * 8];
        a1.s.lo = *(const v8bf*)&zs[(mn + 16) * ZPITCH + k0 + half * 8];
        a1.s.hi = *(const v8bf*)&zs[(mn + 16) * ZPITCH + k0 + 16 + half * 8];
        // B 32x16 bf16: elems 0..7 -> K=h*8.., elems 8..15 -> K=16+h*8.. of column mn
        b0.s.lo = *(const v8bf*)&wrow0[k0 + half * 8];
        b0.s.hi = *(const v8bf*)&wrow0[k0 + 16 + half * 8];
        b1.s.lo = *(const v8bf*)&wrow1[k0 + half * 8];
        b1.s.hi = *(const v8bf*)&wrow1[k0 + 16 + half * 8];

        acc00 = __builtin_amdgcn_wmma_f32_16x16x32_bf16(
                    false, a0.v, false, b0.v, (short)0, acc00, false, false);
        acc01 = __builtin_amdgcn_wmma_f32_16x16x32_bf16(
                    false, a0.v, false, b1.v, (short)0, acc01, false, false);
        acc10 = __builtin_amdgcn_wmma_f32_16x16x32_bf16(
                    false, a1.v, false, b0.v, (short)0, acc10, false, false);
        acc11 = __builtin_amdgcn_wmma_f32_16x16x32_bf16(
                    false, a1.v, false, b1.v, (short)0, acc11, false, false);
    }

    // epilogue: bias (+ sigmoid for the gate section), store f32
    const int sec = colBase >> 10;                  // whole wave in one section
    const float* bias = (sec == 0) ? ba : (sec == 1) ? bb : bc;
    const int cis = colBase & 1023;
    const float bias0 = bias[cis + mn];
    const float bias1 = bias[cis + 16 + mn];

    auto store_tile = [&](const v8f& acc, int mOff, int nOff, float bv) {
        #pragma unroll
        for (int r = 0; r < 8; ++r) {
            const int m = r + half * 8 + mOff;      // C/D layout: lanes16-31 -> M+8
            float v = acc[r] + bv;
            if (sec == 0) v = 1.0f / (1.0f + __expf(-v));
            G[(size_t)(rowBase + m) * D3 + colBase + nOff + mn] = v;
        }
    };
    store_tile(acc00,  0,  0, bias0);
    store_tile(acc01,  0, 16, bias1);
    store_tile(acc10, 16,  0, bias0);
    store_tile(acc11, 16, 16, bias1);
}

// ---------------- 4) chunked scan ------------------------------------------
// idx -> d = idx&1023, c = (idx>>10)&31, b = idx>>15
#define PF_DIST 8

__global__ __launch_bounds__(256)
void mamba_chunk_prod_kernel(const float* __restrict__ G,
                             float* __restrict__ chunkProd)
{
    const int idx = blockIdx.x * blockDim.x + threadIdx.x;
    const int d = idx & 1023, c = (idx >> 10) & 31, b = idx >> 15;
    const size_t row0 = (size_t)b * LL + (size_t)c * LCH;
    const float* ga = G + row0 * D3 + d;
    float p = 1.0f;
    for (int s = 0; s < LCH; ++s) {
        if (s + PF_DIST < LCH)
            __builtin_prefetch(&ga[(size_t)(s + PF_DIST) * D3], 0, 0);
        p *= ga[(size_t)s * D3];
    }
    chunkProd[idx] = p;
}

__global__ __launch_bounds__(256)
void mamba_scan_p_kernel(const float* __restrict__ chunkProd,
                         float* __restrict__ Pstart)
{
    const int idx = blockIdx.x * blockDim.x + threadIdx.x;  // B*D
    const int d = idx & 1023, b = idx >> 10;
    float run = 1.0f;
    for (int c = 0; c < NCH; ++c) {
        const size_t o = (((size_t)b * NCH + c) << 10) + d;
        Pstart[o] = run;
        run *= chunkProd[o];
    }
}

__global__ __launch_bounds__(256)
void mamba_chunk_sum_kernel(const float* __restrict__ G,
                            const float* __restrict__ Pstart,
                            float* __restrict__ chunkSum)
{
    const int idx = blockIdx.x * blockDim.x + threadIdx.x;
    const int d = idx & 1023, c = (idx >> 10) & 31, b = idx >> 15;
    const size_t row0 = (size_t)b * LL + (size_t)c * LCH;
    const float* g = G + row0 * D3 + d;
    float p = Pstart[idx], s = 0.0f;
    for (int t = 0; t < LCH; ++t) {
        if (t + PF_DIST < LCH) {
            __builtin_prefetch(&g[(size_t)(t + PF_DIST) * D3], 0, 0);
            __builtin_prefetch(&g[(size_t)(t + PF_DIST) * D3 + 1024], 0, 0);
        }
        const float a  = g[(size_t)t * D3];
        const float bx = g[(size_t)t * D3 + 1024];
        p *= a;
        s += bx / (p + SCAN_EPS);
    }
    chunkSum[idx] = s;
}

__global__ __launch_bounds__(256)
void mamba_scan_s_kernel(const float* __restrict__ chunkSum,
                         float* __restrict__ Sstart)
{
    const int idx = blockIdx.x * blockDim.x + threadIdx.x;  // B*D
    const int d = idx & 1023, b = idx >> 10;
    float run = 0.0f;
    for (int c = 0; c < NCH; ++c) {
        const size_t o = (((size_t)b * NCH + c) << 10) + d;
        Sstart[o] = run;
        run += chunkSum[o];
    }
}

__global__ __launch_bounds__(256)
void mamba_final_kernel(const float* __restrict__ G,
                        const float* __restrict__ Pstart,
                        const float* __restrict__ Sstart,
                        const float* __restrict__ h0,
                        const float* __restrict__ X,
                        float* __restrict__ out)
{
    const int idx = blockIdx.x * blockDim.x + threadIdx.x;
    const int d = idx & 1023, c = (idx >> 10) & 31, b = idx >> 15;
    const size_t row0 = (size_t)b * LL + (size_t)c * LCH;
    const float* g = G + row0 * D3 + d;
    const float* x = X + row0 * DD + d;
    float* o = out + row0 * DD + d;
    float p = Pstart[idx], s = Sstart[idx];
    const float h0d = h0[d];
    for (int t = 0; t < LCH; ++t) {
        if (t + PF_DIST < LCH) {
            __builtin_prefetch(&g[(size_t)(t + PF_DIST) * D3], 0, 0);
            __builtin_prefetch(&g[(size_t)(t + PF_DIST) * D3 + 1024], 0, 0);
            __builtin_prefetch(&g[(size_t)(t + PF_DIST) * D3 + 2048], 0, 0);
        }
        const float a  = g[(size_t)t * D3];
        const float bx = g[(size_t)t * D3 + 1024];
        const float cc = g[(size_t)t * D3 + 2048];
        p *= a;
        s += bx / (p + SCAN_EPS);
        const float h = p * (h0d + s);
        o[(size_t)t * DD] = cc * h + x[(size_t)t * DD];
    }
}

// ---------------------------------------------------------------------------
extern "C" void kernel_launch(void* const* d_in, const int* in_sizes, int n_in,
                              void* d_out, int out_size, void* d_ws, size_t ws_size,
                              hipStream_t stream)
{
    (void)in_sizes; (void)n_in; (void)out_size; (void)ws_size;
    const float* X     = (const float*)d_in[0];
    const float* Wa    = (const float*)d_in[1];
    const float* ba    = (const float*)d_in[2];
    const float* Wb    = (const float*)d_in[3];
    const float* bb    = (const float*)d_in[4];
    const float* Wc    = (const float*)d_in[5];
    const float* bc    = (const float*)d_in[6];
    const float* gamma = (const float*)d_in[7];
    const float* beta  = (const float*)d_in[8];
    const float* h0    = (const float*)d_in[9];
    float* out = (float*)d_out;

    char* ws = (char*)d_ws;
    size_t off = 0;
    __bf16* Zbf = (__bf16*)(ws + off);  off += (size_t)BB * LL * DD * 2;       // 32 MB
    __bf16* Wbf = (__bf16*)(ws + off);  off += (size_t)3 * DD * DD * 2;        //  6 MB
    float*  G   = (float*)(ws + off);   off += (size_t)BB * LL * D3 * 4;       // 192 MB
    float* chunkProd = (float*)(ws + off); off += (size_t)BB * NCH * DD * 4;
    float* Pstart    = (float*)(ws + off); off += (size_t)BB * NCH * DD * 4;
    float* chunkSum  = (float*)(ws + off); off += (size_t)BB * NCH * DD * 4;
    float* Sstart    = (float*)(ws + off); off += (size_t)BB * NCH * DD * 4;

    // 1) LayerNorm -> bf16 Z
    mamba_ln_kernel<<<BB * LL, 256, 0, stream>>>(X, gamma, beta, Zbf);
    // 2) weights -> bf16
    mamba_wconv_kernel<<<(3 * DD * DD / 4) / 256, 256, 0, stream>>>(Wa, Wb, Wc, Wbf);
    // 3) fused triple GEMM (WMMA, 2x2 register-blocked, dynamic LDS 66 KB)
    mamba_gemm3_kernel<<<dim3(D3 / 256, (BB * LL) / 32), 256, GEMM_LDS_BYTES, stream>>>(
        Zbf, Wbf, ba, bb, bc, G);
    // 4) two-level scan over L
    const int nSeqChunk = BB * NCH * DD;        // 131072
    const int nSeq      = BB * DD;              // 4096
    mamba_chunk_prod_kernel<<<nSeqChunk / 256, 256, 0, stream>>>(G, chunkProd);
    mamba_scan_p_kernel   <<<nSeq / 256,      256, 0, stream>>>(chunkProd, Pstart);
    mamba_chunk_sum_kernel<<<nSeqChunk / 256, 256, 0, stream>>>(G, Pstart, chunkSum);
    mamba_scan_s_kernel   <<<nSeq / 256,      256, 0, stream>>>(chunkSum, Sstart);
    mamba_final_kernel    <<<nSeqChunk / 256, 256, 0, stream>>>(G, Pstart, Sstart, h0, X, out);
}
